// KMeansLayer_66022237274151
// MI455X (gfx1250) — compile-verified, matrix operand order checked by reading, not run on hardware
//
#include <hip/hip_runtime.h>

typedef __attribute__((ext_vector_type(16))) __bf16 v16bf;
typedef __attribute__((ext_vector_type(8)))  __bf16 v8bf;
typedef __attribute__((ext_vector_type(8)))  float  v8f;
typedef __attribute__((ext_vector_type(4)))  unsigned int v4u;
typedef __attribute__((ext_vector_type(4)))  float  v4f;

#define DDIM 256
#define KCLUST 512
#define TILE_M 64
#define XSTRIDE 264   // 256 + 8 ushort pad: 528B row stride -> bank shift of 4/row, conflict-free A-frag loads
#define INV_T 10.0f   // 1 / TEMPERATURE

__device__ __forceinline__ unsigned short f32_to_bf16_rne(float f) {
    unsigned int u = __builtin_bit_cast(unsigned int, f);
    unsigned int r = u + 0x7FFFu + ((u >> 16) & 1u);
    return (unsigned short)(r >> 16);
}
__device__ __forceinline__ float bf16_to_f32(unsigned short h) {
    unsigned int u = (unsigned int)h << 16;
    return __builtin_bit_cast(float, u);
}
__device__ __forceinline__ unsigned int pack2(unsigned short a, unsigned short b) {
    return (unsigned int)a | ((unsigned int)b << 16);
}

// Kernel 1: split centroids into bf16 hi/lo + squared norms. One thread per cluster.
__global__ __launch_bounds__(256) void split_kernel(const float* __restrict__ cent,
                                                    unsigned short* __restrict__ c_hi,
                                                    unsigned short* __restrict__ c_lo,
                                                    float* __restrict__ csq) {
    int k = blockIdx.x * blockDim.x + threadIdx.x;
    if (k >= KCLUST) return;
    const float* c = cent + (size_t)k * DDIM;
    float s = 0.f;
    #pragma unroll 4
    for (int d = 0; d < DDIM; ++d) {
        float f = c[d];
        s += f * f;
        unsigned short hi = f32_to_bf16_rne(f);
        float lo = f - bf16_to_f32(hi);
        c_hi[(size_t)k * DDIM + d] = hi;
        c_lo[(size_t)k * DDIM + d] = f32_to_bf16_rne(lo);
    }
    csq[k] = s;
}

// Kernel 2: fused split-bf16 WMMA GEMM + softmax.
// Block: 256 threads (8 waves), owns 64 rows x 512 cols.
// Wave = (rgrp 0..1: 32 rows) x (cgrp 0..3: 128 cols) -> 2 x 8 tiles of 16x16.
__global__ __launch_bounds__(256) void kmeans_kernel(const float* __restrict__ x,
                                                     const unsigned short* __restrict__ c_hi,
                                                     const unsigned short* __restrict__ c_lo,
                                                     const float* __restrict__ csq,
                                                     float* __restrict__ out) {
    extern __shared__ __align__(16) unsigned char smem[];      // 128 KB dynamic LDS
    unsigned short* xhi = (unsigned short*)smem;               // 33 KB  (GEMM phase)
    unsigned short* xlo = (unsigned short*)(smem + 34816);     // 33 KB  (GEMM phase)
    float* logits = (float*)smem;                              // 128 KB (post-GEMM phase)

    const int tid  = threadIdx.x;
    const int lane = tid & 31;
    const int wave = tid >> 5;
    const int half = lane >> 4;          // K-group select within fragment
    const int r16  = lane & 15;          // M for A-frag, N (cluster) for B-frag
    const int rgrp = wave >> 2;          // 0..1 : rows rgrp*32 .. +31
    const int cgrp = wave & 3;           // 0..3 : cols cgrp*128 .. +127
    const int row_base = blockIdx.x * TILE_M;

    // ---- Stage 64 rows of x into LDS as bf16 hi/lo (each thread: 64 floats) ----
    {
        const int r  = tid >> 2;          // 0..63
        const int d0 = (tid & 3) * 64;    // 0,64,128,192
        const float* src = x + (size_t)(row_base + r) * DDIM + d0;
        #pragma unroll
        for (int i = 0; i < 8; ++i) {     // 8 groups of 8 floats -> one b128 store each
            v4f f0 = *(const v4f*)(src + 8 * i);
            v4f f1 = *(const v4f*)(src + 8 * i + 4);
            unsigned short h[8], l[8];
            #pragma unroll
            for (int j = 0; j < 4; ++j) {
                h[j]     = f32_to_bf16_rne(f0[j]);
                l[j]     = f32_to_bf16_rne(f0[j] - bf16_to_f32(h[j]));
                h[j + 4] = f32_to_bf16_rne(f1[j]);
                l[j + 4] = f32_to_bf16_rne(f1[j] - bf16_to_f32(h[j + 4]));
            }
            v4u hv = { pack2(h[0], h[1]), pack2(h[2], h[3]), pack2(h[4], h[5]), pack2(h[6], h[7]) };
            v4u lv = { pack2(l[0], l[1]), pack2(l[2], l[3]), pack2(l[4], l[5]), pack2(l[6], l[7]) };
            *(v4u*)(xhi + r * XSTRIDE + d0 + 8 * i) = hv;
            *(v4u*)(xlo + r * XSTRIDE + d0 + 8 * i) = lv;
        }
    }
    __syncthreads();

    v8f acc[16] = {};   // [m*8 + t]

    // ---- Split-bf16 GEMM: D = A_hi*B_hi + A_lo*B_hi + A_hi*B_lo ----
    #pragma unroll 1
    for (int kk = 0; kk < DDIM; kk += 32) {
        const int off0 = kk + 8 * half;        // K = 8h + 0..7   (frag VGPR 0-3)
        const int off1 = kk + 16 + 8 * half;   // K = 16+8h+0..7  (frag VGPR 4-7)

        v16bf a_hi[2], a_lo[2];
        #pragma unroll
        for (int m = 0; m < 2; ++m) {
            const int arow = (rgrp * 32 + m * 16 + r16) * XSTRIDE;
            v8bf ah0 = __builtin_bit_cast(v8bf, *(const v4u*)(xhi + arow + off0));
            v8bf ah1 = __builtin_bit_cast(v8bf, *(const v4u*)(xhi + arow + off1));
            v8bf al0 = __builtin_bit_cast(v8bf, *(const v4u*)(xlo + arow + off0));
            v8bf al1 = __builtin_bit_cast(v8bf, *(const v4u*)(xlo + arow + off1));
            a_hi[m] = __builtin_shufflevector(ah0, ah1, 0,1,2,3,4,5,6,7,8,9,10,11,12,13,14,15);
            a_lo[m] = __builtin_shufflevector(al0, al1, 0,1,2,3,4,5,6,7,8,9,10,11,12,13,14,15);
        }

        #pragma unroll
        for (int t = 0; t < 8; ++t) {
            const size_t nrow = (size_t)(cgrp * 128 + t * 16 + r16) * DDIM;
            v8bf bh0 = __builtin_bit_cast(v8bf, *(const v4u*)(c_hi + nrow + off0));
            v8bf bh1 = __builtin_bit_cast(v8bf, *(const v4u*)(c_hi + nrow + off1));
            v8bf bl0 = __builtin_bit_cast(v8bf, *(const v4u*)(c_lo + nrow + off0));
            v8bf bl1 = __builtin_bit_cast(v8bf, *(const v4u*)(c_lo + nrow + off1));
            v16bf b_hi = __builtin_shufflevector(bh0, bh1, 0,1,2,3,4,5,6,7,8,9,10,11,12,13,14,15);
            v16bf b_lo = __builtin_shufflevector(bl0, bl1, 0,1,2,3,4,5,6,7,8,9,10,11,12,13,14,15);

            #pragma unroll
            for (int m = 0; m < 2; ++m) {
                acc[m * 8 + t] = __builtin_amdgcn_wmma_f32_16x16x32_bf16(false, a_hi[m], false, b_hi, (short)0, acc[m * 8 + t], false, false);
                acc[m * 8 + t] = __builtin_amdgcn_wmma_f32_16x16x32_bf16(false, a_lo[m], false, b_hi, (short)0, acc[m * 8 + t], false, false);
                acc[m * 8 + t] = __builtin_amdgcn_wmma_f32_16x16x32_bf16(false, a_hi[m], false, b_lo, (short)0, acc[m * 8 + t], false, false);
            }
        }
    }

    __syncthreads();   // xhi/xlo dead; LDS reused for logits

    // ---- Logits to LDS: logit = (2*cross - ||c||^2)/T  (||x||^2 cancels in softmax) ----
    #pragma unroll
    for (int t = 0; t < 8; ++t) {
        int col = cgrp * 128 + t * 16 + r16;
        float cs = csq[col];
        #pragma unroll
        for (int m = 0; m < 2; ++m) {
            #pragma unroll
            for (int j = 0; j < 8; ++j) {
                int r = rgrp * 32 + m * 16 + j + 8 * half;   // lanes 16-31 hold M = j+8
                logits[r * KCLUST + col] = (2.0f * acc[m * 8 + t][j] - cs) * INV_T;
            }
        }
    }
    __syncthreads();

    // ---- Softmax: wave handles rows wave*8 .. wave*8+7 ----
    #pragma unroll
    for (int rr = 0; rr < 8; ++rr) {
        int r = wave * 8 + rr;
        float vals[16];
        float m = -3.402823466e38f;
        #pragma unroll
        for (int i = 0; i < 16; ++i) {
            vals[i] = logits[r * KCLUST + lane + 32 * i];
            m = fmaxf(m, vals[i]);
        }
        #pragma unroll
        for (int off = 16; off >= 1; off >>= 1)
            m = fmaxf(m, __shfl_xor(m, off, 32));
        float s = 0.f;
        #pragma unroll
        for (int i = 0; i < 16; ++i) { vals[i] = __expf(vals[i] - m); s += vals[i]; }
        #pragma unroll
        for (int off = 16; off >= 1; off >>= 1)
            s += __shfl_xor(s, off, 32);
        float inv = 1.0f / s;
        float* orow = out + (size_t)(row_base + r) * KCLUST;
        #pragma unroll
        for (int i = 0; i < 16; ++i)
            __builtin_nontemporal_store(vals[i] * inv, orow + lane + 32 * i);
    }
}

extern "C" void kernel_launch(void* const* d_in, const int* in_sizes, int n_in,
                              void* d_out, int out_size, void* d_ws, size_t ws_size,
                              hipStream_t stream) {
    const float* x    = (const float*)d_in[0];   // [32,1024,256] fp32
    const float* cent = (const float*)d_in[1];   // [512,256] fp32
    float* out = (float*)d_out;                  // [32,1024,512] fp32

    // Workspace layout: c_hi (256 KB) | c_lo (256 KB) | csq (2 KB)
    unsigned short* c_hi = (unsigned short*)d_ws;
    unsigned short* c_lo = c_hi + (size_t)KCLUST * DDIM;
    float*          csq  = (float*)(c_lo + (size_t)KCLUST * DDIM);

    split_kernel<<<2, 256, 0, stream>>>(cent, c_hi, c_lo, csq);

    const int n_rows = 32 * 1024;                // 32768 samples
    const size_t lds_bytes = 128 * 1024;         // x-staging (padded) union logits
    kmeans_kernel<<<n_rows / TILE_M, 256, lds_bytes, stream>>>(x, c_hi, c_lo, csq, out);
}
